// NALL_16054587752790
// MI455X (gfx1250) — matrix operand order repaired and checked by reading, not compiled, not run on hardware
//
#include <hip/hip_runtime.h>
#include <hip/hip_bf16.h>

typedef _Float16 h4   __attribute__((ext_vector_type(4)));
typedef _Float16 v8h  __attribute__((ext_vector_type(8)));
typedef _Float16 v16h __attribute__((ext_vector_type(16)));
typedef float    v8f  __attribute__((ext_vector_type(8)));
typedef float    f4   __attribute__((ext_vector_type(4)));

#define IN_   256
#define OUT_  256
#define NNODE 512
#define NROWS 384   // B*T
#define GRK   16
#define NRK   8
#define MBLK  64    // rows per m-block
#define NMB   6     // 384 / 64

// LDS partition: Weff [OUT_][IN_] f16 = 128KB, x-stage [MBLK][IN_] f16 = 32KB
#define SMEM_BYTES (OUT_*IN_*2 + MBLK*IN_*2)

__global__ __launch_bounds__(256)
void fused_lora_wmma(const float* __restrict__ x,  const float* __restrict__ W,
                     const float* __restrict__ bv, const float* __restrict__ gA,
                     const float* __restrict__ gB, const float* __restrict__ nA,
                     const float* __restrict__ nB, float* __restrict__ out)
{
    extern __shared__ char smem[];
    _Float16* weff = (_Float16*)smem;                       // [o][k], k contiguous
    _Float16* xs   = (_Float16*)(smem + OUT_*IN_*2);        // [m][k], k contiguous

    const int tid = threadIdx.x;
    const int n   = blockIdx.x;                             // node index

    // ---------------- Phase 1: Weff[o][k] = W[o][k] + (gA gB)/16 + (nA[n] nB[n])/8
    {
        const int k = tid;                                  // 256 threads == IN_
        float ga[GRK];
        #pragma unroll
        for (int r = 0; r < GRK; ++r) ga[r] = gA[k*GRK + r];
        float na[NRK];
        #pragma unroll
        for (int r = 0; r < NRK; ++r) na[r] = nA[((size_t)n*IN_ + k)*NRK + r];

        #pragma unroll 4
        for (int o = 0; o < OUT_; ++o) {
            float acc = W[o*IN_ + k];                       // coalesced across k=tid
            float g = 0.f, p = 0.f;
            #pragma unroll
            for (int r = 0; r < GRK; ++r) g += ga[r] * gB[r*OUT_ + o];   // uniform
            #pragma unroll
            for (int r = 0; r < NRK; ++r) p += na[r] * nB[((size_t)n*NRK + r)*OUT_ + o];
            acc += g * (1.0f/GRK) + p * (1.0f/NRK);
            weff[o*IN_ + k] = (_Float16)acc;                // conflict-free (2B stride)
        }
    }
    __syncthreads();

    const int wid   = tid >> 5;
    const int lane  = tid & 31;
    const int lhalf = lane & 15;
    const int lsel  = lane >> 4;
    const int nbase = wid * 32;                             // 32 output cols per wave

    const float bias0 = bv[nbase + lhalf];
    const float bias1 = bv[nbase + 16 + lhalf];

    const v8f vzero = {0.f,0.f,0.f,0.f,0.f,0.f,0.f,0.f};

    #pragma unroll 1
    for (int mb = 0; mb < NMB; ++mb) {
        // prefetch next m-block: each thread touches one 256B-spaced region
        if (mb + 1 < NMB) {
            const float* pf = x + (((size_t)((mb+1)*MBLK + (tid >> 2)) * NNODE + n) * IN_)
                                + (tid & 3) * 64;
            __builtin_prefetch(pf, 0, 0);                   // global_prefetch_b8
        }

        // ---- stage 64 rows of x into LDS as f16 (4096 float4 loads across 256 threads)
        #pragma unroll 4
        for (int it = 0; it < 16; ++it) {
            const int i   = tid + it * 256;                 // 0..4095
            const int row = i >> 6;                         // 0..63
            const int kq  = i & 63;                         // float4 within row
            const f4 v = ((const f4*)(x + ((size_t)(mb*MBLK + row) * NNODE + n) * IN_))[kq];
            h4 hv;
            hv.x = (_Float16)v.x; hv.y = (_Float16)v.y;
            hv.z = (_Float16)v.z; hv.w = (_Float16)v.w;
            *(h4*)(xs + row*IN_ + kq*4) = hv;
        }
        __syncthreads();

        v8f acc[4][2];
        #pragma unroll
        for (int mt = 0; mt < 4; ++mt)
            #pragma unroll
            for (int nt = 0; nt < 2; ++nt) acc[mt][nt] = vzero;

        // ---- K loop: 8 chunks of K=32, 8 WMMAs per chunk per wave
        #pragma unroll 2
        for (int kc = 0; kc < 8; ++kc) {
            const int kbase = kc * 32;

            // A fragments (16x32 f16): lanes 0-15 row M=lhalf hold K {0..15},
            // lanes 16-31 hold K {16..31}  -> two B128 LDS loads
            v16h a[4];
            #pragma unroll
            for (int mt = 0; mt < 4; ++mt) {
                const _Float16* ap = xs + (mt*16 + lhalf)*IN_ + kbase + lsel*16;
                v8h lo = *(const v8h*)ap;
                v8h hi = *(const v8h*)(ap + 8);
                a[mt] = __builtin_shufflevector(lo, hi, 0,1,2,3,4,5,6,7,
                                                        8,9,10,11,12,13,14,15);
            }

            // B fragments (32x16 f16): lane column = nbase+nt*16+lhalf,
            // lanes 0-15 K=kbase..+15, lanes 16-31 K=kbase+16..+31 -> contiguous 32B
            v16h bm[2];
            #pragma unroll
            for (int nt = 0; nt < 2; ++nt) {
                const _Float16* bp = weff + (nbase + nt*16 + lhalf)*IN_ + kbase + lsel*16;
                bm[nt] = *(const v16h*)bp;
            }

            #pragma unroll
            for (int mt = 0; mt < 4; ++mt)
                #pragma unroll
                for (int nt = 0; nt < 2; ++nt)
                    acc[mt][nt] = __builtin_amdgcn_wmma_f32_16x16x32_f16(
                        false, a[mt], false, bm[nt], (short)0, acc[mt][nt], false, false);
        }

        // ---- writeback: C/D layout lane<16 -> rows 0..7, lane>=16 -> rows 8..15
        #pragma unroll
        for (int mt = 0; mt < 4; ++mt) {
            #pragma unroll
            for (int nt = 0; nt < 2; ++nt) {
                const float badd = nt ? bias1 : bias0;
                const int   col  = nbase + nt*16 + lhalf;
                #pragma unroll
                for (int j = 0; j < 8; ++j) {
                    const int row = mb*MBLK + mt*16 + lsel*8 + j;
                    out[((size_t)row * NNODE + n) * OUT_ + col] = acc[mt][nt][j] + badd;
                }
            }
        }
        __syncthreads();   // LDS reads done before next stage overwrites
    }
}

extern "C" void kernel_launch(void* const* d_in, const int* in_sizes, int n_in,
                              void* d_out, int out_size, void* d_ws, size_t ws_size,
                              hipStream_t stream) {
    (void)in_sizes; (void)n_in; (void)out_size; (void)d_ws; (void)ws_size;
    const float* x  = (const float*)d_in[0];
    const float* W  = (const float*)d_in[1];
    const float* bv = (const float*)d_in[2];
    const float* gA = (const float*)d_in[3];
    const float* gB = (const float*)d_in[4];
    const float* nA = (const float*)d_in[5];
    const float* nB = (const float*)d_in[6];
    float* out = (float*)d_out;

    fused_lora_wmma<<<dim3(NNODE), dim3(256), SMEM_BYTES, stream>>>(
        x, W, bv, gA, gB, nA, nB, out);
}
